// MoeFF_52561809769142
// MI455X (gfx1250) — matrix operand order repaired and checked by reading, not compiled, not run on hardware
//
#include <hip/hip_runtime.h>
#include <hip/hip_bf16.h>

typedef __attribute__((ext_vector_type(16))) __bf16 v16bf;
typedef __attribute__((ext_vector_type(8)))  __bf16 v8bf;
typedef __attribute__((ext_vector_type(8)))  float  v8f;

#define S_TOK 2048
#define DD    1024
#define EE    8
#define MM    4096
#define TT    128    // tokens per tile (tall: weights are the dominant traffic)
#define TN    64     // output cols per tile
#define TK    32     // K per WMMA step (bf16)
#define LC    5      // v8bf chunks per LDS K-row: 32 elems + 8 pad = 80B (16B aligned)
#define HC    9      // v8bf chunks per hst row: 64 elems + 8 pad = 144B (16B aligned)

__device__ __forceinline__ v16bf cat16(v8bf lo, v8bf hi) {
    return __builtin_shufflevector(lo, hi, 0, 1, 2, 3, 4, 5, 6, 7,
                                   8, 9, 10, 11, 12, 13, 14, 15);
}

// pack two fp32 -> packed bf16 pair (lo in [15:0], hi in [31:16]) -> v_cvt_pk_bf16_f32
__device__ __forceinline__ unsigned pk2bf(float lo, float hi) {
    union { __bf16 h[2]; unsigned u; } t;
    t.h[0] = (__bf16)lo; t.h[1] = (__bf16)hi;
    return t.u;
}

// ---------------- init: zero expert counters ----------------
__global__ void moe_init_kernel(int* __restrict__ counts) {
    if (threadIdx.x < EE) counts[threadIdx.x] = 0;
}

// ---------------- cast x -> bf16 ----------------
__global__ void moe_cast_kernel(const float* __restrict__ x, __bf16* __restrict__ xbf, int n) {
    int i = blockIdx.x * blockDim.x + threadIdx.x;
    if (i < n) xbf[i] = (__bf16)x[i];
}

// ---------------- gate: logits -> softmax -> top2 -> compact lists ----------------
__global__ void moe_gate_kernel(const float* __restrict__ x, const float* __restrict__ Wgate,
                                const float* __restrict__ bgate,
                                int* __restrict__ tok_list, int* __restrict__ tk_e,
                                int* __restrict__ tk_pos, float* __restrict__ tk_w,
                                int* __restrict__ counts) {
    int s = blockIdx.x * blockDim.x + threadIdx.x;
    if (s >= S_TOK) return;
    float acc[EE];
#pragma unroll
    for (int e = 0; e < EE; e++) acc[e] = 0.0f;
    const float* xr = x + (size_t)s * DD;
    for (int d = 0; d < DD; d++) {
        float xv = xr[d];
        const float4* wr = (const float4*)(Wgate + (size_t)d * EE);
        float4 w0 = wr[0], w1 = wr[1];
        acc[0] += xv * w0.x; acc[1] += xv * w0.y; acc[2] += xv * w0.z; acc[3] += xv * w0.w;
        acc[4] += xv * w1.x; acc[5] += xv * w1.y; acc[6] += xv * w1.z; acc[7] += xv * w1.w;
    }
#pragma unroll
    for (int e = 0; e < EE; e++) acc[e] += bgate[e];
    float mx = acc[0];
#pragma unroll
    for (int e = 1; e < EE; e++) mx = fmaxf(mx, acc[e]);
    float p[EE], sum = 0.0f;
#pragma unroll
    for (int e = 0; e < EE; e++) { p[e] = __expf(acc[e] - mx); sum += p[e]; }
    // top-2 (strict > keeps lowest index on ties, matching jax.lax.top_k)
    int i0 = 0;
#pragma unroll
    for (int e = 1; e < EE; e++) if (p[e] > p[i0]) i0 = e;
    int i1 = (i0 == 0) ? 1 : 0;
#pragma unroll
    for (int e = 0; e < EE; e++) if (e != i0 && p[e] > p[i1]) i1 = e;
    float w0 = p[i0] / sum, w1 = p[i1] / sum;
    float t = w0 + w1;
    w0 /= t; w1 /= t;
    int pos0 = atomicAdd(&counts[i0], 1);
    tok_list[i0 * S_TOK + pos0] = s;
    tk_e[2 * s] = i0; tk_pos[2 * s] = pos0; tk_w[2 * s] = w0;
    int pos1 = atomicAdd(&counts[i1], 1);
    tok_list[i1 * S_TOK + pos1] = s;
    tk_e[2 * s + 1] = i1; tk_pos[2 * s + 1] = pos1; tk_w[2 * s + 1] = w1;
}

// ---------------- prefix: segment offsets into compacted rows ----------------
__global__ void moe_prefix_kernel(const int* __restrict__ counts, int* __restrict__ segoff) {
    if (threadIdx.x == 0 && blockIdx.x == 0) {
        int off = 0;
        for (int e = 0; e < EE; e++) { segoff[e] = off; off += counts[e]; }
    }
}

// ---------------- ff1: h = swish(x@Wg + bg) * (x@W1 + b1), compacted rows ----------------
struct SM1 {
    union {
        struct { v8bf xs[TT][LC]; v8bf wg[TN][LC]; v8bf w1[TN][LC]; } a;
        v8bf hst[TT][HC];
    } u;
    int toks[TT];
};

__global__ __launch_bounds__(256) void moe_ff1_kernel(
    const __bf16* __restrict__ xbf, const float* __restrict__ Wg, const float* __restrict__ bg,
    const float* __restrict__ W1, const float* __restrict__ b1,
    const int* __restrict__ tok_list, const int* __restrict__ counts,
    const int* __restrict__ segoff, __bf16* __restrict__ hbuf) {
    const int e    = blockIdx.z;
    const int tile = blockIdx.y;
    const int m0   = blockIdx.x * TN;
    const int cnt  = counts[e];
    if (tile * TT >= cnt) return;

    __shared__ SM1 sm;

    const int tid  = threadIdx.x;
    const int lane = tid & 31;
    const int wv   = tid >> 5;   // wave owns token rows [wv*16, wv*16+16)
    const int half = lane >> 4;
    const int l15  = lane & 15;

    // gather rows clamped to a valid row: rows >= cnt only feed discarded C rows
    if (tid < TT) {
        int r = tile * TT + tid;
        sm.toks[tid] = tok_list[e * S_TOK + min(r, cnt - 1)];
    }
    __syncthreads();

    // x gather: 2 chunks/thread: (row, 8-elem chunk) = (tid/4 [+64], tid%4)
    const int gr0 = tid >> 2, gr1 = gr0 + 64;
    const int gch = tid & 3;
    const __bf16* xsrc0 = xbf + (size_t)sm.toks[gr0] * DD + gch * 8;
    const __bf16* xsrc1 = xbf + (size_t)sm.toks[gr1] * DD + gch * 8;

    // W tiles: thread -> (even k-row pair kp, m quad); 2x float4 load, 4x b32 packed store
    const int mq = (tid & 15) * 4;
    const int kp = (tid >> 4) * 2;
    const float* wgp = Wg + (size_t)e * DD * MM + (size_t)kp * MM + m0 + mq;
    const float* w1p = W1 + (size_t)e * DD * MM + (size_t)kp * MM + m0 + mq;

    v8f accA[4], accB[4];
#pragma unroll
    for (int s2 = 0; s2 < 4; s2++) { accA[s2] = (v8f)0.0f; accB[s2] = (v8f)0.0f; }

    for (int k0 = 0; k0 < DD; k0 += TK) {
        __syncthreads();
        sm.u.a.xs[gr0][gch] = *(const v8bf*)(xsrc0 + k0);
        sm.u.a.xs[gr1][gch] = *(const v8bf*)(xsrc1 + k0);

        const float* p0 = wgp + (size_t)k0 * MM;
        float4 g0 = *(const float4*)p0;
        float4 g1 = *(const float4*)(p0 + MM);
        const float* p1 = w1p + (size_t)k0 * MM;
        float4 o0 = *(const float4*)p1;
        float4 o1 = *(const float4*)(p1 + MM);
        ((unsigned*)&sm.u.a.wg[mq + 0][0])[kp >> 1] = pk2bf(g0.x, g1.x);
        ((unsigned*)&sm.u.a.wg[mq + 1][0])[kp >> 1] = pk2bf(g0.y, g1.y);
        ((unsigned*)&sm.u.a.wg[mq + 2][0])[kp >> 1] = pk2bf(g0.z, g1.z);
        ((unsigned*)&sm.u.a.wg[mq + 3][0])[kp >> 1] = pk2bf(g0.w, g1.w);
        ((unsigned*)&sm.u.a.w1[mq + 0][0])[kp >> 1] = pk2bf(o0.x, o1.x);
        ((unsigned*)&sm.u.a.w1[mq + 1][0])[kp >> 1] = pk2bf(o0.y, o1.y);
        ((unsigned*)&sm.u.a.w1[mq + 2][0])[kp >> 1] = pk2bf(o0.z, o1.z);
        ((unsigned*)&sm.u.a.w1[mq + 3][0])[kp >> 1] = pk2bf(o0.w, o1.w);
        __syncthreads();

        // A fragment: chunks [2*half] and [2*half+1] of this wave's token row
        const v8bf* arow = sm.u.a.xs[wv * 16 + l15];
        v16bf afv = cat16(arow[half * 2], arow[half * 2 + 1]);
#pragma unroll
        for (int s2 = 0; s2 < 4; s2++) {
            int nl = s2 * 16 + l15;
            v16bf bgv = cat16(sm.u.a.wg[nl][half * 2], sm.u.a.wg[nl][half * 2 + 1]);
            v16bf b1v = cat16(sm.u.a.w1[nl][half * 2], sm.u.a.w1[nl][half * 2 + 1]);
            accA[s2] = __builtin_amdgcn_wmma_f32_16x16x32_bf16(
                false, afv, false, bgv, (short)0, accA[s2], false, false);
            accB[s2] = __builtin_amdgcn_wmma_f32_16x16x32_bf16(
                false, afv, false, b1v, (short)0, accB[s2], false, false);
        }
    }

    // epilogue: h = swish(a+bg) * (b+b1), staged in LDS, then coalesced b128 stores
    __syncthreads();
#pragma unroll
    for (int s2 = 0; s2 < 4; s2++) {
        int nl = s2 * 16 + l15;
        float bgs = bg[e * MM + m0 + nl];
        float b1s = b1[e * MM + m0 + nl];
#pragma unroll
        for (int v = 0; v < 8; v++) {
            int tl = wv * 16 + v + 8 * half;   // C layout: M = v + 8*half
            float a = accA[s2][v] + bgs;
            float b = accB[s2][v] + b1s;
            ((__bf16*)&sm.u.hst[tl][0])[nl] = (__bf16)((a / (1.0f + __expf(-a))) * b);
        }
    }
    __syncthreads();
    const int base = segoff[e] + tile * TT;
    int rows = cnt - tile * TT; if (rows > TT) rows = TT;
#pragma unroll
    for (int it = 0; it < 4; it++) {
        int cid = tid + it * 256;
        int r = cid >> 3, ch = cid & 7;
        if (r < rows)
            *(v8bf*)&hbuf[(size_t)(base + r) * MM + m0 + ch * 8] = sm.u.hst[r][ch];
    }
}

// ---------------- ff2: y = h @ W2 + b2 per assignment row ----------------
struct SM2 {
    v8bf hs[TT][LC];
    v8bf w2[TN][LC];
};

__global__ __launch_bounds__(256) void moe_ff2_kernel(
    const __bf16* __restrict__ hbuf, const float* __restrict__ W2, const float* __restrict__ b2,
    const int* __restrict__ counts, const int* __restrict__ segoff,
    float* __restrict__ ybuf) {
    const int e    = blockIdx.z;
    const int tile = blockIdx.y;
    const int d0   = blockIdx.x * TN;
    const int cnt  = counts[e];
    if (tile * TT >= cnt) return;

    __shared__ SM2 sm;

    const int tid  = threadIdx.x;
    const int lane = tid & 31;
    const int wv   = tid >> 5;
    const int half = lane >> 4;
    const int l15  = lane & 15;

    const int base = segoff[e] + tile * TT;

    // h gather (rows contiguous); clamp to stay inside written rows
    const int gr0 = tid >> 2, gr1 = gr0 + 64;
    const int gch = tid & 3;
    const int rc0 = min(tile * TT + gr0, cnt - 1) - tile * TT;
    const int rc1 = min(tile * TT + gr1, cnt - 1) - tile * TT;
    const __bf16* hsrc0 = hbuf + (size_t)(base + rc0) * MM + gch * 8;
    const __bf16* hsrc1 = hbuf + (size_t)(base + rc1) * MM + gch * 8;

    const int mq = (tid & 15) * 4;
    const int kp = (tid >> 4) * 2;
    const float* w2p = W2 + (size_t)e * MM * DD + (size_t)kp * DD + d0 + mq;

    v8f acc[4];
#pragma unroll
    for (int s2 = 0; s2 < 4; s2++) acc[s2] = (v8f)0.0f;

    for (int k0 = 0; k0 < MM; k0 += TK) {
        __syncthreads();
        sm.hs[gr0][gch] = *(const v8bf*)(hsrc0 + k0);
        sm.hs[gr1][gch] = *(const v8bf*)(hsrc1 + k0);
        const float* p0 = w2p + (size_t)k0 * DD;
        float4 g0 = *(const float4*)p0;
        float4 g1 = *(const float4*)(p0 + DD);
        ((unsigned*)&sm.w2[mq + 0][0])[kp >> 1] = pk2bf(g0.x, g1.x);
        ((unsigned*)&sm.w2[mq + 1][0])[kp >> 1] = pk2bf(g0.y, g1.y);
        ((unsigned*)&sm.w2[mq + 2][0])[kp >> 1] = pk2bf(g0.z, g1.z);
        ((unsigned*)&sm.w2[mq + 3][0])[kp >> 1] = pk2bf(g0.w, g1.w);
        __syncthreads();

        const v8bf* arow = sm.hs[wv * 16 + l15];
        v16bf afv = cat16(arow[half * 2], arow[half * 2 + 1]);
#pragma unroll
        for (int s2 = 0; s2 < 4; s2++) {
            int nl = s2 * 16 + l15;
            v16bf bwv = cat16(sm.w2[nl][half * 2], sm.w2[nl][half * 2 + 1]);
            acc[s2] = __builtin_amdgcn_wmma_f32_16x16x32_bf16(
                false, afv, false, bwv, (short)0, acc[s2], false, false);
        }
    }
    // write fp32 y rows (per-assignment, combined later; no atomics)
#pragma unroll
    for (int s2 = 0; s2 < 4; s2++) {
        int dg = d0 + s2 * 16 + l15;
        float b2v = b2[e * DD + dg];
#pragma unroll
        for (int v = 0; v < 8; v++) {
            int tl = wv * 16 + v + 8 * half;
            if (tile * TT + tl < cnt)
                ybuf[(size_t)(base + tl) * DD + dg] = acc[s2][v] + b2v;
        }
    }
}

// ---------------- combine: out[s] = w0*y[row0] + w1*y[row1] ----------------
__global__ void moe_combine_kernel(const float* __restrict__ ybuf, const int* __restrict__ tk_e,
                                   const int* __restrict__ tk_pos, const float* __restrict__ tk_w,
                                   const int* __restrict__ segoff, float* __restrict__ out) {
    int i = blockIdx.x * blockDim.x + threadIdx.x;      // over S*DD/4 float4s
    if (i >= S_TOK * DD / 4) return;
    int s = i / (DD / 4), c = i % (DD / 4);
    int r0 = segoff[tk_e[2 * s]]     + tk_pos[2 * s];
    int r1 = segoff[tk_e[2 * s + 1]] + tk_pos[2 * s + 1];
    float w0 = tk_w[2 * s], w1 = tk_w[2 * s + 1];
    float4 a = *((const float4*)(ybuf + (size_t)r0 * DD) + c);
    float4 b = *((const float4*)(ybuf + (size_t)r1 * DD) + c);
    float4 o;
    o.x = w0 * a.x + w1 * b.x;
    o.y = w0 * a.y + w1 * b.y;
    o.z = w0 * a.z + w1 * b.z;
    o.w = w0 * a.w + w1 * b.w;
    ((float4*)out)[i] = o;
}

extern "C" void kernel_launch(void* const* d_in, const int* in_sizes, int n_in,
                              void* d_out, int out_size, void* d_ws, size_t ws_size,
                              hipStream_t stream) {
    const float* x     = (const float*)d_in[0];
    const float* Wgate = (const float*)d_in[1];
    const float* bgate = (const float*)d_in[2];
    const float* Wg    = (const float*)d_in[3];
    const float* bg    = (const float*)d_in[4];
    const float* W1    = (const float*)d_in[5];
    const float* b1    = (const float*)d_in[6];
    const float* W2    = (const float*)d_in[7];
    const float* b2    = (const float*)d_in[8];
    float* out = (float*)d_out;

    char* ws = (char*)d_ws;
    size_t off = 0;
    auto alloc = [&](size_t bytes) -> void* {
        void* p = ws + off;
        off += (bytes + 255) & ~(size_t)255;
        return p;
    };
    __bf16* xbf      = (__bf16*)alloc((size_t)S_TOK * DD * sizeof(__bf16));
    int*    tok_list = (int*)   alloc((size_t)EE * S_TOK * sizeof(int));
    int*    tk_e     = (int*)   alloc((size_t)2 * S_TOK * sizeof(int));
    int*    tk_pos   = (int*)   alloc((size_t)2 * S_TOK * sizeof(int));
    float*  tk_w     = (float*) alloc((size_t)2 * S_TOK * sizeof(float));
    int*    counts   = (int*)   alloc(256);
    int*    segoff   = (int*)   alloc(256);
    __bf16* hbuf     = (__bf16*)alloc((size_t)S_TOK * 2 * MM * sizeof(__bf16));
    float*  ybuf     = (float*) alloc((size_t)S_TOK * 2 * DD * sizeof(float));

    const int n_x = S_TOK * DD;
    moe_init_kernel<<<1, 32, 0, stream>>>(counts);
    moe_cast_kernel<<<(n_x + 255) / 256, 256, 0, stream>>>(x, xbf, n_x);
    moe_gate_kernel<<<(S_TOK + 255) / 256, 256, 0, stream>>>(x, Wgate, bgate, tok_list,
                                                             tk_e, tk_pos, tk_w, counts);
    moe_prefix_kernel<<<1, 32, 0, stream>>>(counts, segoff);
    {
        dim3 grid(MM / TN, S_TOK / TT, EE);   // 64 x 16 x 8, early-exit past counts
        moe_ff1_kernel<<<grid, 256, 0, stream>>>(xbf, Wg, bg, W1, b1,
                                                 tok_list, counts, segoff, hbuf);
    }
    {
        dim3 grid(DD / TN, S_TOK / TT, EE);   // 16 x 16 x 8
        moe_ff2_kernel<<<grid, 256, 0, stream>>>(hbuf, W2, b2, counts, segoff, ybuf);
    }
    moe_combine_kernel<<<(S_TOK * DD / 4 + 255) / 256, 256, 0, stream>>>(
        ybuf, tk_e, tk_pos, tk_w, segoff, out);
}
